// _ObjectAttentionBlock_75093208203699
// MI455X (gfx1250) — compile-verified, matrix operand order checked by reading, not run on hardware
//
#include <hip/hip_runtime.h>
#include <hip/hip_bf16.h>

typedef __attribute__((ext_vector_type(16))) _Float16 v16h;
typedef __attribute__((ext_vector_type(8)))  float    v8f;
typedef __attribute__((ext_vector_type(4)))  unsigned int u32x4;
typedef __attribute__((ext_vector_type(4)))  int      i32x4;
typedef __attribute__((ext_vector_type(8)))  int      i32x8;

#define C_IN 512
#define KC   256
#define MREG 19
#define MP   32
#define HWSZ 16384   // 128*128
#define EPSV 1e-5f
#define SS   33      // padded LDS stride for sim

// 16-bit A/B fragment K index for v16h element i, lane-half hi
__device__ __forceinline__ int kmap(int i, int hi) {
    return ((i >> 3) << 4) + (i & 7) + (hi << 3);
}
// inverse of kmap: for within-32 K offset w, which (element, half) holds it
__device__ __forceinline__ int inv_elem(int w) { return (w & 7) + (((w >> 4) & 1) << 3); }
__device__ __forceinline__ int inv_half(int w) { return (w >> 3) & 1; }

__device__ __forceinline__ v8f vzero8() {
    v8f z;
#pragma unroll
    for (int j = 0; j < 8; ++j) z[j] = 0.0f;
    return z;
}

// ---------------------------------------------------------------------------
// Tensor Data Mover: 1-D async DMA of nelem f16 elements global -> LDS.
// D# per CDNA5 ISA:
//  group0: [1:0]=count=1 | [63:32]=lds_addr | [120:64]=global_addr | [127:126]=2
//  group1: [17:16]=data_size(1 => 2B) | [79:48]=tensor_dim0 | [111:80]=dim1
//          [127:112]=tile_dim0
// ---------------------------------------------------------------------------
__device__ __forceinline__ void tdm_load_1d(unsigned lds_off, const void* gptr,
                                            unsigned nelem) {
    unsigned long long ga = (unsigned long long)(uintptr_t)gptr;
    u32x4 g0;
    g0[0] = 1u;                                            // count=1, user mode
    g0[1] = lds_off;                                       // LDS byte address
    g0[2] = (unsigned)ga;                                  // global addr [31:0]
    g0[3] = (unsigned)((ga >> 32) & 0x1FFFFFFu) | (2u << 30); // [56:32]|type=2
    i32x8 g1;
    g1[0] = 0x00010000;                                    // data_size=2 bytes
    g1[1] = (int)((nelem & 0xFFFFu) << 16);                // tensor_dim0 lo
    g1[2] = (int)((nelem >> 16) & 0xFFFFu) | (1 << 16);    // dim0 hi | dim1=1
    g1[3] = (int)((nelem & 0xFFFFu) << 16);                // tile_dim0
    g1[4] = 0; g1[5] = 0; g1[6] = 0; g1[7] = 0;
    i32x4 gz;
    gz[0] = 0; gz[1] = 0; gz[2] = 0; gz[3] = 0;
#if defined(__clang_major__) && (__clang_major__ >= 23)
    i32x8 gz8;
    gz8[0] = 0; gz8[1] = 0; gz8[2] = 0; gz8[3] = 0;
    gz8[4] = 0; gz8[5] = 0; gz8[6] = 0; gz8[7] = 0;
    __builtin_amdgcn_tensor_load_to_lds(g0, g1, gz, gz, gz8, 0);
#else
    __builtin_amdgcn_tensor_load_to_lds(g0, g1, gz, gz, 0);
#endif
}

__device__ __forceinline__ unsigned lds_offset_of(const void* p) {
    return (unsigned)(uintptr_t)p;
}

// ---------------------------------------------------------------------------
// Kernel 0: fold BN into scale/bias; swizzle Wp / Wu into WMMA B-fragment
// order:  wpf[ks(16)][nt(16)][lane(32)][i(16)],  wuf[nt2(32)][kk(8)][lane][i]
// ---------------------------------------------------------------------------
__global__ __launch_bounds__(256) void prep_kernel(
    const float* __restrict__ Wp, const float* __restrict__ bp,
    const float* __restrict__ gp, const float* __restrict__ betap,
    const float* __restrict__ mp, const float* __restrict__ vp,
    const float* __restrict__ Wu, const float* __restrict__ bu,
    const float* __restrict__ gu, const float* __restrict__ betau,
    const float* __restrict__ mu, const float* __restrict__ vu,
    _Float16* __restrict__ wpf, _Float16* __restrict__ wuf,
    float* __restrict__ sP, float* __restrict__ cP,
    float* __restrict__ sU, float* __restrict__ cU)
{
    int id = blockIdx.x * 256 + threadIdx.x;   // 512 blocks -> 131072 threads
    if (id < C_IN * KC) {
        {   // wpf: B[k=c][n=o] = Wp[o][c]
            int i  = id & 15;
            int ln = (id >> 4) & 31;
            int nt = (id >> 9) & 15;
            int ks = id >> 13;
            int c  = ks * 32 + kmap(i, (ln >> 4) & 1);
            int o  = nt * 16 + (ln & 15);
            wpf[id] = (_Float16)Wp[o * C_IN + c];
        }
        {   // wuf: B[k=kc][n=co] = Wu[co][kc]
            int i   = id & 15;
            int ln  = (id >> 4) & 31;
            int kk  = (id >> 9) & 7;
            int nt2 = id >> 12;
            int kc  = kk * 32 + kmap(i, (ln >> 4) & 1);
            int co  = nt2 * 16 + (ln & 15);
            wuf[id] = (_Float16)Wu[co * KC + kc];
        }
    }
    if (id < KC) {
        float s = gp[id] * rsqrtf(vp[id] + EPSV);
        sP[id] = s;
        cP[id] = bp[id] * s + betap[id] - mp[id] * s;
    }
    if (id < C_IN) {
        float s = gu[id] * rsqrtf(vu[id] + EPSV);
        sU[id] = s;
        cU[id] = bu[id] * s + betau[id] - mu[id] * s;
    }
}

// ---------------------------------------------------------------------------
// Kernel 1: k/v projections of proxy, scattered into B-fragment order.
//   kf[n][kk(8)][t(2)][lane][i] :  B[k=kc][n=m]  = key[kc][m]
//   vf[n][nt(16)][lane][i]      :  B[k=m ][n=kc] = value[m][kc]   (m pad->32)
// ---------------------------------------------------------------------------
__global__ __launch_bounds__(256) void proxy_proj_kernel(
    const float* __restrict__ proxy,
    const float* __restrict__ Wk, const float* __restrict__ bk,
    const float* __restrict__ gk, const float* __restrict__ bek,
    const float* __restrict__ mk, const float* __restrict__ vk,
    const float* __restrict__ Wv, const float* __restrict__ bv,
    const float* __restrict__ gv, const float* __restrict__ bev,
    const float* __restrict__ mv, const float* __restrict__ vv,
    _Float16* __restrict__ kf, _Float16* __restrict__ vf)
{
    int n = blockIdx.x;
    int o = threadIdx.x;                       // output channel (0..255)
    float sK = gk[o] * rsqrtf(vk[o] + EPSV);
    float cK = bk[o] * sK + bek[o] - mk[o] * sK;
    float sV = gv[o] * rsqrtf(vv[o] + EPSV);
    float cV = bv[o] * sV + bev[o] - mv[o] * sV;
    const float* wkr = Wk + (size_t)o * C_IN;
    const float* wvr = Wv + (size_t)o * C_IN;
    const float* px  = proxy + (size_t)n * C_IN * MREG;

    int kkk = o >> 5, wk_ = o & 31;
    int khi = inv_half(wk_), kel = inv_elem(wk_);
    int vnt = o >> 4, vlr = o & 15;

    for (int m = 0; m < MP; ++m) {
        float rk = 0.0f, rv = 0.0f;
        if (m < MREG) {
            float ak = 0.0f, av = 0.0f;
            for (int c = 0; c < C_IN; ++c) {
                float p = px[c * MREG + m];
                ak += wkr[c] * p;
                av += wvr[c] * p;
            }
            rk = ak * sK + cK; rk = rk > 0.0f ? rk : 0.0f;
            rv = av * sV + cV; rv = rv > 0.0f ? rv : 0.0f;
        }
        int t = m >> 4, lrm = m & 15;
        kf[((((size_t)n * 8 + kkk) * 2 + t) * 32 + khi * 16 + lrm) * 16 + kel] = (_Float16)rk;
        vf[(((size_t)n * 16 + vnt) * 32 + inv_half(m) * 16 + vlr) * 16 + inv_elem(m)] = (_Float16)rv;
    }
}

// ---------------------------------------------------------------------------
// Kernel 2: fused q-proj -> sim -> softmax -> context -> f_up (WMMA f16/f32)
// Block = 64 threads (2 waves), tile = 32 pixels. grid = (HW/32, N)
// Weight B-fragments are streamed global->LDS by the Tensor Data Mover,
// double-buffered, one DMA in flight (s_wait_tensorcnt 1).
// ---------------------------------------------------------------------------
__global__ __launch_bounds__(64) void fused_attn_kernel(
    const float*    __restrict__ x,
    const _Float16* __restrict__ wpf,
    const _Float16* __restrict__ wuf,
    const _Float16* __restrict__ kf,
    const _Float16* __restrict__ vf,
    const float* __restrict__ sP, const float* __restrict__ cP,
    const float* __restrict__ sU, const float* __restrict__ cU,
    float* __restrict__ out)
{
    __shared__ alignas(32) _Float16 wstage[2 * 8192];     // TDM dbl buffer, 32 KB
    __shared__ alignas(32) _Float16 qf[2 * 8 * 32 * 16];  // q/ctx A-frags, 16 KB
    __shared__ alignas(32) _Float16 pf[2 * 32 * 16];      // prob A-frags,   2 KB
    __shared__ float sb[32 * SS];                         // sim f32,      4.1 KB

    const int tid  = threadIdx.x;
    const int lane = tid & 31;
    const int wv   = tid >> 5;              // wave 0/1 -> rows 16w..16w+15
    const int hi   = lane >> 4;
    const int lr   = lane & 15;
    const int n    = blockIdx.y;
    const int s0   = blockIdx.x * 32;

    const v16h* kfv  = (const v16h*)kf + (size_t)n * 8 * 2 * 32;
    const v16h* vfv  = (const v16h*)vf + (size_t)n * 16 * 32;
    const v16h* qfv  = (const v16h*)qf;
    const v16h* pfv  = (const v16h*)pf;
    const unsigned ws0 = lds_offset_of(&wstage[0]);
    const unsigned ws1 = lds_offset_of(&wstage[8192]);

    // -------- Phase A: q[32,256] = relu(BN(x . Wp^T)) -> qf (A-frag order) --
    {
        v8f acc[16];
#pragma unroll
        for (int t = 0; t < 16; ++t) acc[t] = vzero8();

        if (wv == 0) tdm_load_1d(ws0, wpf, 8192);          // chunk ks=0

        const float* xb = x + (size_t)n * C_IN * HWSZ + s0 + wv * 16 + lr;
#pragma unroll 1
        for (int ks = 0; ks < 16; ++ks) {           // K = 512 in steps of 32
            if (wv == 0) {
                if (ks < 15) {
                    tdm_load_1d((ks & 1) ? ws0 : ws1, wpf + (ks + 1) * 8192, 8192);
                    __builtin_amdgcn_s_wait_tensorcnt(1);  // chunk ks complete
                } else {
                    __builtin_amdgcn_s_wait_tensorcnt(0);
                }
            }
            __syncthreads();                                // publish chunk ks

            if (ks + 1 < 16)
                __builtin_prefetch(xb + (size_t)((ks + 1) * 32) * HWSZ, 0, 0);
            v16h a;
#pragma unroll
            for (int i = 0; i < 16; ++i)
                a[i] = (_Float16)xb[(size_t)(ks * 32 + kmap(i, hi)) * HWSZ];

            const v16h* wb = (const v16h*)&wstage[(ks & 1) * 8192];
#pragma unroll
            for (int nt = 0; nt < 16; ++nt) {       // 256 out channels
                v16h b = wb[nt * 32 + lane];
                acc[nt] = __builtin_amdgcn_wmma_f32_16x16x32_f16(
                    false, a, false, b, (short)0, acc[nt], false, false);
            }
            __syncthreads();             // reads done before buffer is re-filled
        }
#pragma unroll
        for (int nt = 0; nt < 16; ++nt) {
            int o  = nt * 16 + lr;
            int kk = o >> 5, w = o & 31;
            int h2 = inv_half(w), i2 = inv_elem(w);
            float s = sP[o], c0 = cP[o];
#pragma unroll
            for (int r = 0; r < 8; ++r) {
                int row = r + 8 * hi;
                float y = acc[nt][r] * s + c0;
                y = y > 0.0f ? y : 0.0f;
                qf[(((wv * 8 + kk) * 32) + h2 * 16 + row) * 16 + i2] = (_Float16)y;
            }
        }
    }
    __syncthreads();

    // -------- Phase B: sim[32,19] = (q . k) * Kc^-0.5 --------
    {
        v16h aq[8];
#pragma unroll
        for (int kk = 0; kk < 8; ++kk) aq[kk] = qfv[(wv * 8 + kk) * 32 + lane];

        v8f sa0 = vzero8(), sa1 = vzero8();
#pragma unroll
        for (int kk = 0; kk < 8; ++kk) {            // K = 256 in steps of 32
            v16h b0 = kfv[(kk * 2 + 0) * 32 + lane];
            v16h b1 = kfv[(kk * 2 + 1) * 32 + lane];
            sa0 = __builtin_amdgcn_wmma_f32_16x16x32_f16(false, aq[kk], false, b0, (short)0, sa0, false, false);
            sa1 = __builtin_amdgcn_wmma_f32_16x16x32_f16(false, aq[kk], false, b1, (short)0, sa1, false, false);
        }
        const float sc = 0.0625f;                    // 256^-0.5
#pragma unroll
        for (int r = 0; r < 8; ++r) {
            int row = wv * 16 + r + 8 * hi;
            sb[row * SS + lr]      = sa0[r] * sc;
            sb[row * SS + 16 + lr] = sa1[r] * sc;
        }
    }
    __syncthreads();

    // -------- softmax over M=19 (one thread per pixel row) -> pf frags -----
    if (tid < 32) {
        int row = tid, wv2 = row >> 4, lr2 = row & 15;
        float mx = -1e30f;
#pragma unroll
        for (int m = 0; m < MREG; ++m) mx = fmaxf(mx, sb[row * SS + m]);
        float e[MREG];
        float sum = 0.0f;
#pragma unroll
        for (int m = 0; m < MREG; ++m) { e[m] = __expf(sb[row * SS + m] - mx); sum += e[m]; }
        float inv = 1.0f / sum;
#pragma unroll
        for (int m = 0; m < MP; ++m) {
            float p = (m < MREG) ? e[m] * inv : 0.0f;
            pf[((wv2 * 32) + inv_half(m) * 16 + lr2) * 16 + inv_elem(m)] = (_Float16)p;
        }
    }
    __syncthreads();

    // -------- Phase C: ctx[32,256] = P . V (K = 32 padded) -> qf frags -----
    {
        v16h ap = pfv[wv * 32 + lane];
#pragma unroll
        for (int nt = 0; nt < 16; ++nt) {
            v16h b = vfv[nt * 32 + lane];
            v8f cz = vzero8();
            cz = __builtin_amdgcn_wmma_f32_16x16x32_f16(false, ap, false, b, (short)0, cz, false, false);
            int o  = nt * 16 + lr;
            int kk = o >> 5, w = o & 31;
            int h2 = inv_half(w), i2 = inv_elem(w);
#pragma unroll
            for (int r = 0; r < 8; ++r) {
                int row = r + 8 * hi;
                qf[(((wv * 8 + kk) * 32) + h2 * 16 + row) * 16 + i2] = (_Float16)cz[r];
            }
        }
    }
    __syncthreads();

    // -------- Phase D: out[32,512] = relu(BN(ctx . Wu^T)) --------
    {
        v16h ac[8];
#pragma unroll
        for (int kk = 0; kk < 8; ++kk) ac[kk] = qfv[(wv * 8 + kk) * 32 + lane];

        if (wv == 0) tdm_load_1d(ws0, wuf, 4096);          // chunk nt2=0

        const size_t outb = (size_t)n * C_IN * HWSZ + s0;
#pragma unroll 1
        for (int nt2 = 0; nt2 < 32; ++nt2) {        // 512 out channels
            if (wv == 0) {
                if (nt2 < 31) {
                    tdm_load_1d((nt2 & 1) ? ws0 : ws1, wuf + (nt2 + 1) * 4096, 4096);
                    __builtin_amdgcn_s_wait_tensorcnt(1);
                } else {
                    __builtin_amdgcn_s_wait_tensorcnt(0);
                }
            }
            __syncthreads();

            const v16h* wb = (const v16h*)&wstage[(nt2 & 1) * 8192];
            v8f acc = vzero8();
#pragma unroll
            for (int kk = 0; kk < 8; ++kk) {        // K = 256
                v16h b = wb[kk * 32 + lane];
                acc = __builtin_amdgcn_wmma_f32_16x16x32_f16(false, ac[kk], false, b, (short)0, acc, false, false);
            }
            int o = nt2 * 16 + lr;
            float s = sU[o], c0 = cU[o];
#pragma unroll
            for (int r = 0; r < 8; ++r) {
                float y = acc[r] * s + c0;
                y = y > 0.0f ? y : 0.0f;
                out[outb + (size_t)o * HWSZ + wv * 16 + r + 8 * hi] = y;
            }
            __syncthreads();
        }
    }
}

// ---------------------------------------------------------------------------
extern "C" void kernel_launch(void* const* d_in, const int* in_sizes, int n_in,
                              void* d_out, int out_size, void* d_ws, size_t ws_size,
                              hipStream_t stream) {
    const float* x     = (const float*)d_in[0];
    const float* proxy = (const float*)d_in[1];
    // f_pixel
    const float* Wp = (const float*)d_in[2];  const float* bp  = (const float*)d_in[3];
    const float* gp = (const float*)d_in[4];  const float* bep = (const float*)d_in[5];
    const float* mp = (const float*)d_in[6];  const float* vp  = (const float*)d_in[7];
    // f_object (keys)
    const float* Wk = (const float*)d_in[8];  const float* bk  = (const float*)d_in[9];
    const float* gk = (const float*)d_in[10]; const float* bek = (const float*)d_in[11];
    const float* mk = (const float*)d_in[12]; const float* vk  = (const float*)d_in[13];
    // f_down (values)
    const float* Wv = (const float*)d_in[14]; const float* bv  = (const float*)d_in[15];
    const float* gv = (const float*)d_in[16]; const float* bev = (const float*)d_in[17];
    const float* mv = (const float*)d_in[18]; const float* vv  = (const float*)d_in[19];
    // f_up
    const float* Wu = (const float*)d_in[20]; const float* bu  = (const float*)d_in[21];
    const float* gu = (const float*)d_in[22]; const float* beu = (const float*)d_in[23];
    const float* mu = (const float*)d_in[24]; const float* vu  = (const float*)d_in[25];

    float* out = (float*)d_out;

    char* ws = (char*)d_ws;
    _Float16* wpf = (_Float16*)(ws);                 // 512*256 f16 = 256 KB
    _Float16* wuf = (_Float16*)(ws + 262144);        // 256*512 f16 = 256 KB
    _Float16* kf  = (_Float16*)(ws + 524288);        // 8*8192  f16 = 128 KB
    _Float16* vf  = (_Float16*)(ws + 655360);        // 8*8192  f16 = 128 KB
    float* sP = (float*)(ws + 786432);
    float* cP = sP + KC;
    float* sU = cP + KC;
    float* cU = sU + C_IN;

    prep_kernel<<<512, 256, 0, stream>>>(Wp, bp, gp, bep, mp, vp,
                                         Wu, bu, gu, beu, mu, vu,
                                         wpf, wuf, sP, cP, sU, cU);

    proxy_proj_kernel<<<8, 256, 0, stream>>>(proxy,
                                             Wk, bk, gk, bek, mk, vk,
                                             Wv, bv, gv, bev, mv, vv,
                                             kf, vf);

    fused_attn_kernel<<<dim3(HWSZ / 32, 8), 64, 0, stream>>>(
        x, wpf, wuf, kf, vf, sP, cP, sU, cU, out);
}
